// AllOrderAttention_89979564851540
// MI455X (gfx1250) — matrix operand order repaired and checked by reading, not compiled, not run on hardware
//
#include <hip/hip_runtime.h>
#include <cstddef>
#include <cstdint>

#define DEVI __device__ __forceinline__

typedef __attribute__((ext_vector_type(16))) _Float16 v16h;
typedef __attribute__((ext_vector_type(8)))  _Float16 v8h;
typedef __attribute__((ext_vector_type(8)))  float    v8f;
typedef __attribute__((ext_vector_type(4)))  float    v4f;

#define SQRT3F 1.7320508075688772f
#define NNODE  10000
#define NK     160000   // N*K

DEVI v8f v8f_zero() { v8f z = {0.f,0.f,0.f,0.f,0.f,0.f,0.f,0.f}; return z; }
DEVI v8h v8h_zero() {
  v8h z = {(_Float16)0.f,(_Float16)0.f,(_Float16)0.f,(_Float16)0.f,
           (_Float16)0.f,(_Float16)0.f,(_Float16)0.f,(_Float16)0.f};
  return z;
}
DEVI float sigmoid_f(float x) { return 1.f / (1.f + __expf(-x)); }

// ---------------------------------------------------------------------------
// Wave-level 64 x TN GEMM micro-kernel: f16 A (LDS) x f16 B (global, [TN][TK]
// transposed layout) -> f32 accumulators, via V_WMMA_F32_16X16X32_F16.
// Wave w of 8: M-tile m0=(w>>1)*16, N-half nbase=(w&1)*(TN/2), NT=TN/32 tiles.
// A frag (16x32 f16): lane l16 holds row M=l16; halves = 16 consecutive K
// (K+16*hi), matching the B fragment order.
// B frag (32x16 f16): lane holds col N=l16, halves = 16 consecutive K
// (K+16*hi) -> one contiguous 32B load from the transposed weight.
// ---------------------------------------------------------------------------
template<int TK, int TN>
DEVI void wave_gemm(const _Float16* As, const _Float16* __restrict__ WT,
                    v8f* acc, int m0, int nbase, int l16, int hi)
{
  constexpr int NT = TN / 32;
#pragma unroll
  for (int k = 0; k < TK; k += 32) {
    const _Float16* ap = As + (m0 + l16) * TK + k + hi * 16;
    v8h alo = *(const v8h*)(ap);
    v8h ahi = *(const v8h*)(ap + 8);
    v16h a = __builtin_shufflevector(alo, ahi, 0,1,2,3,4,5,6,7,8,9,10,11,12,13,14,15);
#pragma unroll
    for (int j = 0; j < NT; ++j) {
      int col = nbase + j * 16 + l16;
      v16h b = *(const v16h*)(WT + (size_t)col * TK + (k + hi * 16));
      acc[j] = __builtin_amdgcn_wmma_f32_16x16x32_f16(false, a, false, b,
                                                      (short)0, acc[j], false, false);
    }
  }
}

// ---------------------------------------------------------------------------
// Weight transpose + f32 -> f16 conversion into the weight pool.
// ---------------------------------------------------------------------------
struct TDesc { const float* src; int K; int N; int off; };
struct TArr  { TDesc d[32]; };

__global__ __launch_bounds__(256) void transpose_to_f16(TArr ta, _Float16* __restrict__ wp)
{
  TDesc t = ta.d[blockIdx.x];
  int total = t.K * t.N;
  for (int i = threadIdx.x; i < total; i += 256) {
    int k = i / t.N, n = i % t.N;
    wp[(size_t)t.off + (size_t)n * t.K + k] = (_Float16)t.src[i];
  }
}

// ---------------------------------------------------------------------------
// Gate: sigmoid(edge_feature @ pe_W + pe_b + node_irreps[:,0,:] @ ns_W + ns_b)
// ---------------------------------------------------------------------------
__global__ __launch_bounds__(256) void gate_kernel(
    const float* __restrict__ edge_feature, const float* __restrict__ nir,
    const _Float16* __restrict__ peWT, const _Float16* __restrict__ nsWT,
    const float* __restrict__ pe_b, const float* __restrict__ ns_b,
    float* __restrict__ gate, int Nn)
{
  __shared__ alignas(16) _Float16 As[64 * 128];
  int tid = threadIdx.x, blk = blockIdx.x;
  int wave = tid >> 5, lane = tid & 31;
  int m0 = (wave >> 1) * 16, nbase = (wave & 1) * 128;
  int l16 = lane & 15, hi = lane >> 4;
  v8f acc[8];
#pragma unroll
  for (int j = 0; j < 8; ++j) acc[j] = v8f_zero();

  // pass 1: A = edge_feature [N,128]
  for (int i = tid; i < 64 * 32; i += 256) {
    int r = i / 32, c = i % 32;
    int g = blk * 64 + r;
    v4f t = {0.f,0.f,0.f,0.f};
    if (g < Nn) t = *(const v4f*)(edge_feature + (size_t)g * 128 + c * 4);
#pragma unroll
    for (int q = 0; q < 4; ++q) As[r * 128 + c * 4 + q] = (_Float16)t[q];
  }
  __syncthreads();
  wave_gemm<128, 256>(As, peWT, acc, m0, nbase, l16, hi);
  __syncthreads();
  // pass 2: A = node_irreps_input[:,0,:] (row stride 9*128)
  for (int i = tid; i < 64 * 32; i += 256) {
    int r = i / 32, c = i % 32;
    int g = blk * 64 + r;
    v4f t = {0.f,0.f,0.f,0.f};
    if (g < Nn) t = *(const v4f*)(nir + (size_t)g * 1152 + c * 4);
#pragma unroll
    for (int q = 0; q < 4; ++q) As[r * 128 + c * 4 + q] = (_Float16)t[q];
  }
  __syncthreads();
  wave_gemm<128, 256>(As, nsWT, acc, m0, nbase, l16, hi);

#pragma unroll
  for (int j = 0; j < 8; ++j) {
    int col = nbase + j * 16 + l16;
    float bv = pe_b[col] + ns_b[col];
#pragma unroll
    for (int v = 0; v < 8; ++v) {
      int row = m0 + v + (hi ? 8 : 0);
      int g = blk * 64 + row;
      if (g < Nn) gate[(size_t)g * 256 + col] = sigmoid_f(acc[j][v] + bv);
    }
  }
}

// ---------------------------------------------------------------------------
// Fused 3-MLP radial kernel: one x_edge tile staged once, 9 GEMM stages.
// ---------------------------------------------------------------------------
struct RadOff { int o[9]; };
struct RadB   { const float* b[9]; };

template<int TK, int TN>
DEVI void mlp_stage(const _Float16* As, const _Float16* WT, const float* bias,
                    _Float16* dstL, _Float16* dstG, int growBase, bool act,
                    int wave, int lane)
{
  constexpr int NT = TN / 32;
  int m0 = (wave >> 1) * 16;
  int nbase = (wave & 1) * (TN / 2);
  int l16 = lane & 15, hi = lane >> 4;
  v8f acc[NT];
#pragma unroll
  for (int j = 0; j < NT; ++j) acc[j] = v8f_zero();
  wave_gemm<TK, TN>(As, WT, acc, m0, nbase, l16, hi);
#pragma unroll
  for (int j = 0; j < NT; ++j) {
    int col = nbase + j * 16 + l16;
    float bv = bias ? bias[col] : 0.f;
#pragma unroll
    for (int v = 0; v < 8; ++v) {
      int row = m0 + v + (hi ? 8 : 0);
      float x = acc[j][v] + bv;
      if (act) x = x * sigmoid_f(x);           // SiLU
      if (dstL) dstL[row * TN + col] = (_Float16)x;
      else      dstG[(size_t)(growBase + row) * TN + col] = (_Float16)x;
    }
  }
}

__global__ __launch_bounds__(256) void fused_radial_mlp(
    const float* __restrict__ x_edge, const _Float16* __restrict__ wp,
    RadOff ro, RadB rb,
    _Float16* __restrict__ ih0, _Float16* __restrict__ ih1, _Float16* __restrict__ ih2)
{
  extern __shared__ __align__(16) _Float16 smem[];
  _Float16* xs = smem;               // 64*384 f16
  _Float16* h1 = smem + 64 * 384;    // 64*128 f16
  _Float16* h2 = h1 + 64 * 128;      // 64*128 f16
  int tid = threadIdx.x, blk = blockIdx.x;
  int wave = tid >> 5, lane = tid & 31;
  int growBase = blk * 64;           // 2500 * 64 == 160000 exactly

  for (int i = tid; i < 64 * 96; i += 256) {
    int r = i / 96, c = i % 96;
    v4f t = *(const v4f*)(x_edge + (size_t)(growBase + r) * 384 + c * 4);
#pragma unroll
    for (int q = 0; q < 4; ++q) xs[r * 384 + c * 4 + q] = (_Float16)t[q];
  }
  __syncthreads();
  // rad0: 384->128->128->128
  mlp_stage<384,128>(xs, wp + ro.o[0], rb.b[0], h1, nullptr, 0, true,  wave, lane); __syncthreads();
  mlp_stage<128,128>(h1, wp + ro.o[1], rb.b[1], h2, nullptr, 0, true,  wave, lane); __syncthreads();
  mlp_stage<128,128>(h2, wp + ro.o[2], rb.b[2], nullptr, ih0, growBase, false, wave, lane); __syncthreads();
  // rad1: 384->128->128->64
  mlp_stage<384,128>(xs, wp + ro.o[3], rb.b[3], h1, nullptr, 0, true,  wave, lane); __syncthreads();
  mlp_stage<128,128>(h1, wp + ro.o[4], rb.b[4], h2, nullptr, 0, true,  wave, lane); __syncthreads();
  mlp_stage<128, 64>(h2, wp + ro.o[5], rb.b[5], nullptr, ih1, growBase, false, wave, lane); __syncthreads();
  // rad2: 384->128->128->32
  mlp_stage<384,128>(xs, wp + ro.o[6], rb.b[6], h1, nullptr, 0, true,  wave, lane); __syncthreads();
  mlp_stage<128,128>(h1, wp + ro.o[7], rb.b[7], h2, nullptr, 0, true,  wave, lane); __syncthreads();
  mlp_stage<128, 32>(h2, wp + ro.o[8], rb.b[8], nullptr, ih2, growBase, false, wave, lane);
}

// ---------------------------------------------------------------------------
// Generic SO3 per-degree linear: rows of A are (node, comp) with comp in
// [seg, seg+dim_l).  Epilogue modes:
//   0: out f16 plain (+bias)        1: d_out  = x * gate[:, :128]
//   2: d_out += x * gate[:, 128:]   3: d_out += x * (1 - gate[:, 128:])
// ---------------------------------------------------------------------------
template<int TK, int TN, bool AHALF>
__global__ __launch_bounds__(256) void so3_kernel(
    const void* __restrict__ Ain, const _Float16* __restrict__ WT,
    const float* __restrict__ bias,
    _Float16* __restrict__ outH, float* __restrict__ outF,
    const float* __restrict__ gate, int mode,
    int dim_l, int seg_off, int Mrows)
{
  __shared__ alignas(16) _Float16 As[64 * TK];
  int tid = threadIdx.x, blk = blockIdx.x;
  if (AHALF) {
    const _Float16* A = (const _Float16*)Ain;
    for (int i = tid; i < 64 * (TK / 8); i += 256) {
      int r = i / (TK / 8), c = i % (TK / 8);
      int g = blk * 64 + r;
      v8h t = v8h_zero();
      if (g < Mrows) {
        int node = g / dim_l, comp = seg_off + (g - node * dim_l);
        t = *(const v8h*)(A + ((size_t)node * 9 + comp) * TK + c * 8);
      }
      *(v8h*)(As + r * TK + c * 8) = t;
    }
  } else {
    const float* A = (const float*)Ain;
    for (int i = tid; i < 64 * (TK / 4); i += 256) {
      int r = i / (TK / 4), c = i % (TK / 4);
      int g = blk * 64 + r;
      v4f t = {0.f,0.f,0.f,0.f};
      if (g < Mrows) {
        int node = g / dim_l, comp = seg_off + (g - node * dim_l);
        t = *(const v4f*)(A + ((size_t)node * 9 + comp) * TK + c * 4);
      }
#pragma unroll
      for (int q = 0; q < 4; ++q) As[r * TK + c * 4 + q] = (_Float16)t[q];
    }
  }
  __syncthreads();

  int wave = tid >> 5, lane = tid & 31;
  int m0 = (wave >> 1) * 16, nbase = (wave & 1) * (TN / 2);
  int l16 = lane & 15, hi = lane >> 4;
  constexpr int NT = TN / 32;
  v8f acc[NT];
#pragma unroll
  for (int j = 0; j < NT; ++j) acc[j] = v8f_zero();
  wave_gemm<TK, TN>(As, WT, acc, m0, nbase, l16, hi);

#pragma unroll
  for (int j = 0; j < NT; ++j) {
    int col = nbase + j * 16 + l16;
    float bv = bias ? bias[col] : 0.f;
#pragma unroll
    for (int v = 0; v < 8; ++v) {
      int row = m0 + v + (hi ? 8 : 0);
      int g = blk * 64 + row;
      if (g >= Mrows) continue;
      int node = g / dim_l, comp = seg_off + (g - node * dim_l);
      float x = acc[j][v] + bv;
      size_t o = ((size_t)node * 9 + comp) * TN + col;
      if (mode == 0)      outH[o] = (_Float16)x;
      else if (mode == 1) outF[o]  = x * gate[(size_t)node * 256 + col];
      else if (mode == 2) outF[o] += x * gate[(size_t)node * 256 + 128 + col];
      else                outF[o] += x * (1.f - gate[(size_t)node * 256 + 128 + col]);
    }
  }
}

// ---------------------------------------------------------------------------
// Aggregation (gather) kernels.  value / V1 / V2 are L2-resident (~63MB).
// ---------------------------------------------------------------------------
__global__ __launch_bounds__(256) void agg0_kernel(
    const float* __restrict__ alpha, const _Float16* __restrict__ ih0,
    const int* __restrict__ idxs, const float* __restrict__ value,
    _Float16* __restrict__ msg0, int Nn)
{
  int t = blockIdx.x * 256 + threadIdx.x;
  int node = t >> 7, c = t & 127;
  if (node >= Nn) return;
  float acc[9] = {0.f,0.f,0.f,0.f,0.f,0.f,0.f,0.f,0.f};
  for (int k = 0; k < 16; ++k) {
    int e = node * 16 + k;
    float w = alpha[(size_t)e * 8 + (c >> 4)] * (float)ih0[(size_t)e * 128 + c];
    int id = idxs[e];
    const float* vb = value + (size_t)id * 9 * 128 + c;
#pragma unroll
    for (int m = 0; m < 9; ++m) acc[m] += w * vb[m * 128];
  }
#pragma unroll
  for (int m = 0; m < 9; ++m)
    msg0[((size_t)node * 9 + m) * 128 + c] = (_Float16)acc[m];
}

__global__ __launch_bounds__(256) void agg1_kernel(
    const float* __restrict__ alpha, const _Float16* __restrict__ ih1,
    const int* __restrict__ idxe, const float* __restrict__ node_pos,
    const float* __restrict__ edge_dis, const _Float16* __restrict__ V1,
    _Float16* __restrict__ msg1, int Nn)
{
  int t = blockIdx.x * 256 + threadIdx.x;
  int node = t >> 6, c = t & 63;
  if (node >= Nn) return;
  float px = node_pos[node * 3 + 0], py = node_pos[node * 3 + 1], pz = node_pos[node * 3 + 2];
  float acc[9] = {0.f,0.f,0.f,0.f,0.f,0.f,0.f,0.f,0.f};
  for (int k = 0; k < 16; ++k) {
    int e = node * 16 + k;
    int id = idxe[e];
    float w = alpha[(size_t)e * 8 + (c >> 3)] * (float)ih1[(size_t)e * 64 + c]
              / (edge_dis[e] + 1e-8f);
    float rx = px - node_pos[id * 3 + 0];
    float ry = py - node_pos[id * 3 + 1];
    float rz = pz - node_pos[id * 3 + 2];
    const _Float16* vb = V1 + (size_t)id * 9 * 64 + c;
    float v0 = (float)vb[0];
#pragma unroll
    for (int m = 0; m < 9; ++m) acc[m] += w * (float)vb[m * 64];
    acc[1] += w * v0 * rx;  // Y_1 = (x, y, z) into components 1..3
    acc[2] += w * v0 * ry;
    acc[3] += w * v0 * rz;
  }
#pragma unroll
  for (int m = 0; m < 9; ++m)
    msg1[((size_t)node * 9 + m) * 64 + c] = (_Float16)acc[m];
}

__global__ __launch_bounds__(256) void agg2_kernel(
    const float* __restrict__ alpha, const _Float16* __restrict__ ih2,
    const int* __restrict__ idxe, const float* __restrict__ node_pos,
    const float* __restrict__ edge_dis, const _Float16* __restrict__ V2,
    _Float16* __restrict__ msg2, int Nn)
{
  int t = blockIdx.x * 256 + threadIdx.x;
  int node = t >> 5, c = t & 31;
  if (node >= Nn) return;
  float px = node_pos[node * 3 + 0], py = node_pos[node * 3 + 1], pz = node_pos[node * 3 + 2];
  float acc[9] = {0.f,0.f,0.f,0.f,0.f,0.f,0.f,0.f,0.f};
  for (int k = 0; k < 16; ++k) {
    int e = node * 16 + k;
    int id = idxe[e];
    float d = edge_dis[e];
    float w = alpha[(size_t)e * 8 + (c >> 2)] * (float)ih2[(size_t)e * 32 + c]
              / (d * d + 1e-8f);
    float rx = px - node_pos[id * 3 + 0];
    float ry = py - node_pos[id * 3 + 1];
    float rz = pz - node_pos[id * 3 + 2];
    float y0 = SQRT3F * rx * ry;
    float y1 = SQRT3F * ry * rz;
    float y2 = 0.5f * (2.f * rz * rz - rx * rx - ry * ry);
    float y3 = SQRT3F * rx * rz;
    float y4 = 0.5f * SQRT3F * (rx * rx - ry * ry);
    const _Float16* vb = V2 + (size_t)id * 9 * 32 + c;
    float v0 = (float)vb[0];
#pragma unroll
    for (int m = 0; m < 9; ++m) acc[m] += w * (float)vb[m * 32];
    acc[4] += w * v0 * y0;  // Y_2 into components 4..8
    acc[5] += w * v0 * y1;
    acc[6] += w * v0 * y2;
    acc[7] += w * v0 * y3;
    acc[8] += w * v0 * y4;
  }
#pragma unroll
  for (int m = 0; m < 9; ++m)
    msg2[((size_t)node * 9 + m) * 32 + c] = (_Float16)acc[m];
}

// ---------------------------------------------------------------------------
// Host side.
// ---------------------------------------------------------------------------
extern "C" void kernel_launch(void* const* d_in, const int* in_sizes, int n_in,
                              void* d_out, int out_size, void* d_ws, size_t ws_size,
                              hipStream_t stream)
{
  (void)in_sizes; (void)out_size; (void)ws_size;
  // ---- input leaf mapping (43 leaves if pytree-flattened; 31 dict entries) ----
  const void* L[43];
  if (n_in >= 43) {
    for (int i = 0; i < 43; ++i) L[i] = d_in[i];
  } else {
    for (int i = 0; i < 9; ++i) L[i] = d_in[i];
    const float* w;
    w = (const float*)d_in[9];  L[9]  = w; L[10] = w + 384*128; L[11] = w + 384*128 + 128*128;
    w = (const float*)d_in[10]; L[12] = w; L[13] = w + 128;     L[14] = w + 256;
    w = (const float*)d_in[11]; L[15] = w; L[16] = w + 384*128; L[17] = w + 384*128 + 128*128;
    w = (const float*)d_in[12]; L[18] = w; L[19] = w + 128;     L[20] = w + 256;
    w = (const float*)d_in[13]; L[21] = w; L[22] = w + 384*128; L[23] = w + 384*128 + 128*128;
    w = (const float*)d_in[14]; L[24] = w; L[25] = w + 128;     L[26] = w + 256;
    for (int i = 0; i < 16; ++i) L[27 + i] = d_in[15 + i];
  }
  const float* alpha        = (const float*)L[0];
  const float* value        = (const float*)L[1];
  const float* x_edge       = (const float*)L[2];
  const float* node_pos     = (const float*)L[3];
  const float* edge_dis     = (const float*)L[4];
  const float* edge_feature = (const float*)L[5];
  const float* nir          = (const float*)L[6];
  const int*   idx_node     = (const int*)L[7];
  const int*   idx_exp      = (const int*)L[8];
  const float* radW[9] = { (const float*)L[9],  (const float*)L[10], (const float*)L[11],
                           (const float*)L[15], (const float*)L[16], (const float*)L[17],
                           (const float*)L[21], (const float*)L[22], (const float*)L[23] };
  const float* radBp[9] = { (const float*)L[12], (const float*)L[13], (const float*)L[14],
                            (const float*)L[18], (const float*)L[19], (const float*)L[20],
                            (const float*)L[24], (const float*)L[25], (const float*)L[26] };
  const float* pzW  = (const float*)L[27]; const float* pzB  = (const float*)L[28];
  const float* pvfW = (const float*)L[29]; const float* pvfB = (const float*)L[30];
  const float* pfW  = (const float*)L[31]; const float* pfB  = (const float*)L[32];
  const float* pvsW = (const float*)L[33]; const float* pvsB = (const float*)L[34];
  const float* psW  = (const float*)L[35]; const float* psB  = (const float*)L[36];
  const float* tfW  = (const float*)L[37]; const float* tsW  = (const float*)L[38];
  const float* peW  = (const float*)L[39]; const float* peB  = (const float*)L[40];
  const float* nsW  = (const float*)L[41]; const float* nsB  = (const float*)L[42];

  // ---- workspace partition (bytes, 256-aligned) ----
  char* wsb = (char*)d_ws;
  size_t off = 0;
  auto ball = [&](size_t bytes) { size_t r = off; off += (bytes + 255) & ~(size_t)255; return r; };
  size_t wpB  = ball((size_t)512000 * 2);         // f16 weight pool
  size_t ih0B = ball((size_t)NK * 128 * 2);
  size_t ih1B = ball((size_t)NK * 64 * 2);
  size_t ih2B = ball((size_t)NK * 32 * 2);
  size_t V1B  = ball((size_t)NNODE * 9 * 64 * 2);
  size_t V2B  = ball((size_t)NNODE * 9 * 32 * 2);
  size_t m0B  = ball((size_t)NNODE * 9 * 128 * 2);
  size_t m1B  = ball((size_t)NNODE * 9 * 64 * 2);
  size_t m2B  = ball((size_t)NNODE * 9 * 32 * 2);
  size_t t1B  = ball((size_t)NNODE * 9 * 128 * 2);
  size_t t2B  = ball((size_t)NNODE * 9 * 128 * 2);
  size_t gtB  = ball((size_t)NNODE * 256 * 4);
  _Float16* wp   = (_Float16*)(wsb + wpB);
  _Float16* ih0  = (_Float16*)(wsb + ih0B);
  _Float16* ih1  = (_Float16*)(wsb + ih1B);
  _Float16* ih2  = (_Float16*)(wsb + ih2B);
  _Float16* V1h  = (_Float16*)(wsb + V1B);
  _Float16* V2h  = (_Float16*)(wsb + V2B);
  _Float16* msg0 = (_Float16*)(wsb + m0B);
  _Float16* msg1 = (_Float16*)(wsb + m1B);
  _Float16* msg2 = (_Float16*)(wsb + m2B);
  _Float16* t1h  = (_Float16*)(wsb + t1B);
  _Float16* t2h  = (_Float16*)(wsb + t2B);
  float*    gateF = (float*)(wsb + gtB);
  float*    outF  = (float*)d_out;

  // ---- weight pool offsets (in halves) + transpose descriptors ----
  int wo = 0;
  auto walloc = [&](int k, int n) { int r = wo; wo += k * n; return r; };
  TArr ta; int nd = 0;
  auto add = [&](const float* s, int k, int n) { int o = walloc(k, n);
                                                 ta.d[nd].src = s; ta.d[nd].K = k;
                                                 ta.d[nd].N = n; ta.d[nd].off = o;
                                                 ++nd; return o; };
  RadOff ro; RadB rb;
  static const int rk[9] = {384,128,128, 384,128,128, 384,128,128};
  static const int rn[9] = {128,128,128, 128,128, 64, 128,128, 32};
  for (int i = 0; i < 9; ++i) { ro.o[i] = add(radW[i], rk[i], rn[i]); rb.b[i] = radBp[i]; }
  int o_pz[3], o_pvf[3], o_pf[3], o_pvs[3], o_ps[3], o_tf[3], o_ts[3];
  for (int l = 0; l < 3; ++l) o_pz[l]  = add(pzW  + (size_t)l * 128 * 128, 128, 128);
  for (int l = 0; l < 3; ++l) o_pvf[l] = add(pvfW + (size_t)l * 128 *  64, 128,  64);
  for (int l = 0; l < 3; ++l) o_pf[l]  = add(pfW  + (size_t)l * 128 * 128, 128, 128);
  for (int l = 0; l < 3; ++l) o_pvs[l] = add(pvsW + (size_t)l * 128 *  32, 128,  32);
  for (int l = 0; l < 3; ++l) o_ps[l]  = add(psW  + (size_t)l * 128 * 128, 128, 128);
  for (int l = 0; l < 3; ++l) o_tf[l]  = add(tfW  + (size_t)l *  64 * 128,  64, 128);
  for (int l = 0; l < 3; ++l) o_ts[l]  = add(tsW  + (size_t)l *  32 * 128,  32, 128);
  int o_pe = add(peW, 128, 256);
  int o_ns = add(nsW, 128, 256);

  transpose_to_f16<<<nd, 256, 0, stream>>>(ta, wp);

  // ---- gate ----
  gate_kernel<<<(NNODE + 63) / 64, 256, 0, stream>>>(
      edge_feature, nir, wp + o_pe, wp + o_ns, peB, nsB, gateF, NNODE);

  // ---- fused radial MLPs (x_edge staged once; 81920B dynamic LDS) ----
  fused_radial_mlp<<<NK / 64, 256, 81920, stream>>>(x_edge, wp, ro, rb, ih0, ih1, ih2);

  static const int dimL[3] = {1, 3, 5}, segL[3] = {0, 1, 4};
  auto grid_l = [&](int l) { return (NNODE * dimL[l] + 63) / 64; };

  // ---- V1 = so3(value, pv_fir), V2 = so3(value, pv_sec) ----
  for (int l = 0; l < 3; ++l) {
    int M = NNODE * dimL[l];
    so3_kernel<128, 64, false><<<grid_l(l), 256, 0, stream>>>(
        value, wp + o_pvf[l], l == 0 ? pvfB : nullptr, V1h, nullptr, nullptr, 0,
        dimL[l], segL[l], M);
    so3_kernel<128, 32, false><<<grid_l(l), 256, 0, stream>>>(
        value, wp + o_pvs[l], l == 0 ? pvsB : nullptr, V2h, nullptr, nullptr, 0,
        dimL[l], segL[l], M);
  }

  // ---- order 0 ----
  agg0_kernel<<<(NNODE * 128 + 255) / 256, 256, 0, stream>>>(
      alpha, ih0, idx_node, value, msg0, NNODE);
  for (int l = 0; l < 3; ++l)
    so3_kernel<128, 128, true><<<grid_l(l), 256, 0, stream>>>(
        msg0, wp + o_pz[l], l == 0 ? pzB : nullptr, nullptr, outF, gateF, 1,
        dimL[l], segL[l], NNODE * dimL[l]);

  // ---- order 1 ----
  agg1_kernel<<<(NNODE * 64 + 255) / 256, 256, 0, stream>>>(
      alpha, ih1, idx_exp, node_pos, edge_dis, V1h, msg1, NNODE);
  for (int l = 0; l < 3; ++l)
    so3_kernel<64, 128, true><<<grid_l(l), 256, 0, stream>>>(
        msg1, wp + o_tf[l], nullptr, t1h, nullptr, nullptr, 0,
        dimL[l], segL[l], NNODE * dimL[l]);
  for (int l = 0; l < 3; ++l)
    so3_kernel<128, 128, true><<<grid_l(l), 256, 0, stream>>>(
        t1h, wp + o_pf[l], l == 0 ? pfB : nullptr, nullptr, outF, gateF, 2,
        dimL[l], segL[l], NNODE * dimL[l]);

  // ---- order 2 ----
  agg2_kernel<<<(NNODE * 32 + 255) / 256, 256, 0, stream>>>(
      alpha, ih2, idx_exp, node_pos, edge_dis, V2h, msg2, NNODE);
  for (int l = 0; l < 3; ++l)
    so3_kernel<32, 128, true><<<grid_l(l), 256, 0, stream>>>(
        msg2, wp + o_ts[l], nullptr, t2h, nullptr, nullptr, 0,
        dimL[l], segL[l], NNODE * dimL[l]);
  for (int l = 0; l < 3; ++l)
    so3_kernel<128, 128, true><<<grid_l(l), 256, 0, stream>>>(
        t2h, wp + o_ps[l], l == 0 ? psB : nullptr, nullptr, outF, gateF, 3,
        dimL[l], segL[l], NNODE * dimL[l]);
}
